// LSTM_86440511800183
// MI455X (gfx1250) — matrix operand order repaired and checked by reading, not compiled, not run on hardware
//
#include <hip/hip_runtime.h>

typedef __attribute__((ext_vector_type(16))) _Float16 v16h;
typedef __attribute__((ext_vector_type(8)))  _Float16 v8h;
typedef __attribute__((ext_vector_type(8)))  float    v8f;

#define T_LEN 8192
#define HID   40
#define NTHR  320   // 10 waves; wave w owns hidden units 4w..4w+3 (all 4 gates each)

#define LOG2E 1.4426950408889634f

static __device__ __forceinline__ float tanh_fast(float x) {
#if __has_builtin(__builtin_amdgcn_tanhf)
    return __builtin_amdgcn_tanhf(x);                       // v_tanh_f32
#elif __has_builtin(__builtin_amdgcn_tanh_f32)
    return __builtin_amdgcn_tanh_f32(x);
#else
    const float e = __builtin_amdgcn_exp2f(-2.0f * LOG2E * x);
    return __builtin_fmaf(2.0f, __builtin_amdgcn_rcpf(1.0f + e), -1.0f);
#endif
}

// Broadcast the value held by lane (lane & ~3) + POS to all 4 lanes of the quad.
// DPP16 quad_perm (pure VALU) when available; ds_swizzle fallback otherwise.
template <int POS>
static __device__ __forceinline__ float quad_bcast(float v) {
    const int av = __float_as_int(v);
#if __has_builtin(__builtin_amdgcn_mov_dpp)
    // quad_perm:[POS,POS,POS,POS] -> ctrl = POS * 0b01010101
    return __int_as_float(__builtin_amdgcn_mov_dpp(av, POS * 0x55, 0xf, 0xf, true));
#else
    return __int_as_float(__builtin_amdgcn_ds_swizzle(av, 0x8000 | (POS * 0x55)));
#endif
}

__global__ __launch_bounds__(NTHR, 1)
void lstm_wmma_kernel(const float* __restrict__ x_seq,
                      const float* __restrict__ W_ih,
                      const float* __restrict__ W_hh,
                      const float* __restrict__ b_ih,
                      const float* __restrict__ b_hh,
                      const float* __restrict__ W_lin,
                      const float* __restrict__ b_lin,
                      float* __restrict__ out) {
    __shared__ float xs[T_LEN];                       // whole input sequence
    __shared__ __align__(16) _Float16 h16[2][64];     // double-buffered h (f16), pad 40..63 = 0

    const int tid  = threadIdx.x;
    const int w    = tid >> 5;         // wave id 0..9
    const int lane = tid & 31;
    const int n    = lane & 15;
    const int hi   = lane >> 4;        // K-group select within A/B tiles
    // Permuted gate-row assignment: WMMA output column n of wave w computes
    // permuted row r = 16w + n  ==  unit j = 4w + (n>>2), gate q = n&3.
    const int q    = n & 3;            // 0=i, 1=f, 2=g, 3=o
    const int j    = 4 * w + (n >> 2); // hidden unit owned by this lane
    const int row  = q * HID + j;      // row in the reference [4H, ...] layout

    // Stage input sequence into LDS (once).
    for (int i = tid; i < T_LEN; i += NTHR) xs[i] = x_seq[i];
    if (tid < 128) ((_Float16*)h16)[tid] = (_Float16)0.0f;  // h0 = 0, both buffers

    // Per-lane constants. Activation: act = amul2 * tanh(ascl3 * gpre) + aadd2
    //   sigmoid (q!=2): 0.5 + 0.5*tanh(0.5*x)
    //   tanh    (q==2): tanh(x)
    const float ascl3 = (q == 2) ? 1.0f : 0.5f;
    const float amul2 = (q == 2) ? 1.0f : 0.5f;
    const float aadd2 = (q == 2) ? 0.0f : 0.5f;
    const float wih3  = W_ih[row] * ascl3;          // pre-scaled input weight
    const float bias  = b_ih[row] + b_hh[row];
    v8f biasvec;                                    // WMMA C operand carries the bias
    #pragma unroll
    for (int k = 0; k < 8; ++k) biasvec[k] = bias;

    // Build B operands once: B[k][n] = W_hh[row(n)][k], K padded 40->64 with zeros.
    // 16-bit B 32x16 k order used here (same as A): lanes 0-15 <-> K=0..15, lanes 16-31 <-> K=16..31,
    // half jj <-> K = 16*hi + jj, N = lane&15.
    v16h B0, B1;
    const int kbase = hi * 16;
    #pragma unroll
    for (int jj = 0; jj < 16; ++jj) {
        const int k0 = kbase + jj;          // 0..31
        B0[jj] = (_Float16)W_hh[row * HID + k0];
        const int k1 = 32 + kbase + jj;     // 32..63
        B1[jj] = (k1 < HID) ? (_Float16)W_hh[row * HID + k1] : (_Float16)0.0f;
    }

    float c = 0.0f;          // cell state for unit j (meaningful only in q==0 lanes)
    float xnext = 0.0f;      // software-pipelined x[t]

    __syncthreads();
    xnext = xs[0];

    // A-matrix (16-bit 16x32) K offsets: halves 0..7 <-> K=16*hi+jj,
    // halves 8..15 <-> K=16*hi+8+(jj-8)
    const int o0 = hi * 16;
    const int o1 = hi * 16 + 8;

    auto step = [&](const _Float16* rb, _Float16* wb, int t) {
        // Issue A loads first (they gate the WMMAs).
        v8h a0lo = *(const v8h*)&rb[o0];
        v8h a0hi = *(const v8h*)&rb[o1];
        v8h a1lo = *(const v8h*)&rb[o0 + 32];   // zeros when K>=40
        v8h a1hi = *(const v8h*)&rb[o1 + 32];   // always zeros (K>=40)
        const float xm = xnext * wih3;          // off critical path
        v16h A0, A1;
        #pragma unroll
        for (int jj = 0; jj < 8; ++jj) {
            A0[jj] = a0lo[jj]; A0[8 + jj] = a0hi[jj];
            A1[jj] = a1lo[jj]; A1[8 + jj] = a1hi[jj];
        }

        // Two independent WMMAs; acc0's C operand carries the bias splat.
        v8f acc1 = {};
        acc1 = __builtin_amdgcn_wmma_f32_16x16x32_f16(false, A1, false, B1,
                                                      (short)0, acc1, false, false);
        v8f acc0 = __builtin_amdgcn_wmma_f32_16x16x32_f16(false, A0, false, B0,
                                                          (short)0, biasvec, false, false);

        // gpre (pre-scaled) -> activation: add -> fma -> tanh -> fma
        const float arg = __builtin_fmaf(ascl3, acc0[0] + acc1[0], xm);
        const float act = __builtin_fmaf(amul2, tanh_fast(arg), aadd2);

        // In-wave gather: the q==0 lane already holds gi; broadcast f/g/o via DPP.
        const float gf = quad_bcast<1>(act);
        const float gg = quad_bcast<2>(act);
        const float go = quad_bcast<3>(act);

        // Cell + hidden update (valid in q==0 lanes; others compute dead values).
        c = __builtin_fmaf(gf, c, act * gg);
        if (q == 0 && lane < 16) {
            wb[j] = (_Float16)(go * tanh_fast(c));
        }

        // Split barrier: signal, fetch next x under the barrier wait, then wait.
        __asm__ volatile("s_wait_dscnt 0x0\n\ts_barrier_signal -1" ::: "memory");
        xnext = xs[(t + 1) & (T_LEN - 1)];
        __asm__ volatile("s_barrier_wait -1" ::: "memory");
    };

    #pragma unroll 1
    for (int t = 0; t < T_LEN; t += 2) {
        step(&h16[0][0], &h16[1][0], t);       // even step: read buf0, write buf1
        step(&h16[1][0], &h16[0][0], t + 1);   // odd step:  read buf1, write buf0
    }

    // Final linear: out = h_T . W_lin + b_lin. Last write (t=8191) -> buffer 0.
    if (tid == 0) {
        float s = b_lin[0];
        #pragma unroll
        for (int k = 0; k < HID; ++k) s += (float)h16[0][k] * W_lin[k];
        out[0] = s;
    }
}

extern "C" void kernel_launch(void* const* d_in, const int* in_sizes, int n_in,
                              void* d_out, int out_size, void* d_ws, size_t ws_size,
                              hipStream_t stream) {
    (void)in_sizes; (void)n_in; (void)out_size; (void)d_ws; (void)ws_size;
    const float* x    = (const float*)d_in[0];
    const float* Wih  = (const float*)d_in[1];
    const float* Whh  = (const float*)d_in[2];
    const float* bih  = (const float*)d_in[3];
    const float* bhh  = (const float*)d_in[4];
    const float* Wlin = (const float*)d_in[5];
    const float* blin = (const float*)d_in[6];
    float* outp = (float*)d_out;
    hipLaunchKernelGGL(lstm_wmma_kernel, dim3(1), dim3(NTHR), 0, stream,
                       x, Wih, Whh, bih, bhh, Wlin, blin, outp);
}
